// GPRGNN_85985245266264
// MI455X (gfx1250) — compile-verified, matrix operand order checked by reading, not run on hardware
//
#include <hip/hip_runtime.h>
#include <hip/hip_bf16.h>

typedef __attribute__((ext_vector_type(16))) __bf16 v16bf;
typedef __attribute__((ext_vector_type(8)))  float  v8f;

#define F_IN 512
#define HID  256
#define NCLS 32
#define KHOP 10

// ---------------- utility kernels ----------------

__global__ __launch_bounds__(256) void fill_f32(float* __restrict__ p, float v, long long n) {
    long long t = (long long)blockIdx.x * blockDim.x + threadIdx.x;
    if (t < n) p[t] = v;
}

// Repack a KxN f32 weight matrix into WMMA-fragment-major bf16:
// out[((ct*(K/32) + kt)*32 + lane)*16 + j] = (bf16)W[k(j,lane)*N + ct*16 + idx(lane)]
// so each lane's 16-element B fragment is 32 contiguous bytes (one v16bf load).
__global__ __launch_bounds__(256) void pack_b_frag(const float* __restrict__ W, __bf16* __restrict__ out,
                                                   int K, int N) {
    int t = blockIdx.x * blockDim.x + threadIdx.x;
    if (t >= K * N) return;
    const int ktPerCt = K / 32;
    int ct   = t / (512 * ktPerCt);
    int rem  = t % (512 * ktPerCt);
    int kt   = rem >> 9;           // /512
    int r2   = rem & 511;
    int lane = r2 >> 4;
    int j    = r2 & 15;
    int half = lane >> 4;
    int idx  = lane & 15;
    int i    = j >> 1;
    int lo   = j & 1;
    int k    = kt * 32 + ((i >> 2) << 4) + half * 8 + ((i & 3) << 1) + lo;
    int col  = ct * 16 + idx;
    out[t] = (__bf16)W[(long long)k * N + col];
}

// ---------------- GCN norm ----------------

__global__ __launch_bounds__(256) void degree_kernel(const int* __restrict__ ei, float* __restrict__ deg, int nE) {
    int e = blockIdx.x * blockDim.x + threadIdx.x;
    if (e < nE) atomicAdd(&deg[ei[nE + e]], 1.0f);   // dst degrees (self-loops via deg init = 1)
}

__global__ __launch_bounds__(256) void dinv_kernel(const float* __restrict__ deg, float* __restrict__ dinv, int n) {
    int i = blockIdx.x * blockDim.x + threadIdx.x;
    if (i < n) { float d = deg[i]; dinv[i] = d > 0.f ? rsqrtf(d) : 0.f; }
}

__global__ __launch_bounds__(256) void norm_kernel(const int* __restrict__ ei, const float* __restrict__ dinv,
                                                   float* __restrict__ norm, int nE) {
    int e = blockIdx.x * blockDim.x + threadIdx.x;
    if (e < nE) norm[e] = dinv[ei[e]] * dinv[ei[nE + e]];
}

// ---------------- fused MLP: h = relu(x@W1+b1)@W2+b2 via WMMA bf16 ----------------
// 128 threads (4 waves) per block, 16 rows of x per block.
// B matrices are pre-packed fragment-major: one v16bf (32B) load per fragment.

__global__ __launch_bounds__(128) void mlp_fused(const float* __restrict__ x,
                                                 const __bf16* __restrict__ w1p, const float* __restrict__ b1,
                                                 const __bf16* __restrict__ w2p, const float* __restrict__ b2,
                                                 float* __restrict__ h, int nRows) {
    __shared__ __bf16 As[16 * F_IN];   // 16 KB : 16 x 512 input tile (bf16)
    __shared__ __bf16 H1s[16 * HID];   //  8 KB : 16 x 256 hidden tile (bf16)

    const int row0 = blockIdx.x * 16;
    const int tid  = threadIdx.x;
    const int wave = tid >> 5;
    const int lane = tid & 31;
    const int half = lane >> 4;
    const int idx  = lane & 15;

    // Stage A tile (f32 -> bf16) into LDS.
    for (int i = tid; i < 16 * F_IN; i += 128) {
        int r = i / F_IN, c = i % F_IN;
        int gr = row0 + r;
        float v = (gr < nRows) ? x[(long long)gr * F_IN + c] : 0.0f;
        As[i] = (__bf16)v;
    }
    __syncthreads();

    // ---- GEMM1: (16 x 512) @ (512 x 256), each wave does 4 of 16 col tiles ----
    for (int t = 0; t < 4; ++t) {
        const int ct   = wave + 4 * t;
        const int col0 = ct * 16;
        v8f acc = {};
        const v16bf* bptr = (const v16bf*)(w1p + (((size_t)ct * (F_IN / 32)) * 32 + lane) * 16);
        for (int kt = 0; kt < F_IN / 32; ++kt) {
            const int kb = kt * 32;
            v16bf a;
#pragma unroll
            for (int i = 0; i < 8; ++i) {
                int k0 = kb + ((i >> 2) << 4) + half * 8 + ((i & 3) << 1);
                a[2 * i]     = As[idx * F_IN + k0];
                a[2 * i + 1] = As[idx * F_IN + k0 + 1];
            }
            v16bf b = bptr[kt * 32];   // fragment-major: stride 32 lanes of 16 elems
            acc = __builtin_amdgcn_wmma_f32_16x16x32_bf16(false, a, false, b, (short)0, acc, false, false);
        }
#pragma unroll
        for (int j = 0; j < 8; ++j) {
            int m = half * 8 + j;               // C/D row = j + half*8
            int c = col0 + idx;
            float v = acc[j] + b1[c];
            H1s[m * HID + c] = (__bf16)(v > 0.f ? v : 0.f);
        }
    }
    __syncthreads();

    // ---- GEMM2: (16 x 256) @ (256 x 32), waves 0/1 ----
    if (wave < 2) {
        const int col0 = wave * 16;
        v8f acc = {};
        const v16bf* bptr = (const v16bf*)(w2p + (((size_t)wave * (HID / 32)) * 32 + lane) * 16);
        for (int kt = 0; kt < HID / 32; ++kt) {
            const int kb = kt * 32;
            v16bf a;
#pragma unroll
            for (int i = 0; i < 8; ++i) {
                int k0 = kb + ((i >> 2) << 4) + half * 8 + ((i & 3) << 1);
                a[2 * i]     = H1s[idx * HID + k0];
                a[2 * i + 1] = H1s[idx * HID + k0 + 1];
            }
            v16bf b = bptr[kt * 32];
            acc = __builtin_amdgcn_wmma_f32_16x16x32_bf16(false, a, false, b, (short)0, acc, false, false);
        }
#pragma unroll
        for (int j = 0; j < 8; ++j) {
            int m  = half * 8 + j;
            int gr = row0 + m;
            if (gr < nRows) h[(long long)gr * NCLS + col0 + idx] = acc[j] + b2[col0 + idx];
        }
    }
}

// ---------------- propagation ----------------

__global__ __launch_bounds__(256) void init_acc_kernel(const float* __restrict__ h, float* __restrict__ hk,
                                                       float* __restrict__ acc, const float* __restrict__ gamma,
                                                       long long n) {
    long long t = (long long)blockIdx.x * blockDim.x + threadIdx.x;
    if (t < n) { float v = h[t]; hk[t] = v; acc[t] = gamma[0] * v; }
}

// one wave per edge, lanes = 32 contiguous features: coalesced gather + L2 atomics
__global__ __launch_bounds__(256) void edge_prop_kernel(const int* __restrict__ ei, const float* __restrict__ norm,
                                                        const float* __restrict__ hk, float* __restrict__ hn, int nE) {
    long long t = (long long)blockIdx.x * blockDim.x + threadIdx.x;
    long long e = t >> 5;
    int f = (int)(t & 31);
    if (e >= nE) return;
    int s = ei[e];
    int d = ei[nE + e];
    float v = norm[e] * hk[(long long)s * NCLS + f];
    atomicAdd(&hn[(long long)d * NCLS + f], v);
}

// add self-loop term, accumulate gamma[k] * h_k
__global__ __launch_bounds__(256) void finish_prop_kernel(float* __restrict__ hn, const float* __restrict__ hk,
                                                          const float* __restrict__ dinv, float* __restrict__ acc,
                                                          const float* __restrict__ gamma, int k, long long n) {
    long long t = (long long)blockIdx.x * blockDim.x + threadIdx.x;
    if (t >= n) return;
    long long i = t >> 5;
    float di = dinv[i];
    float v = hn[t] + di * di * hk[t];
    hn[t] = v;
    acc[t] += gamma[k] * v;
}

// ---------------- log-softmax: one wave32 per row, lane == class ----------------

__global__ __launch_bounds__(256) void logsoftmax_kernel(const float* __restrict__ acc, float* __restrict__ out, int n) {
    int row  = blockIdx.x * (blockDim.x >> 5) + (threadIdx.x >> 5);
    int lane = threadIdx.x & 31;
    if (row >= n) return;
    float v = acc[(long long)row * NCLS + lane];
    float m = v;
#pragma unroll
    for (int o = 16; o > 0; o >>= 1) m = fmaxf(m, __shfl_xor(m, o, 32));
    float e = __expf(v - m);
    float s = e;
#pragma unroll
    for (int o = 16; o > 0; o >>= 1) s += __shfl_xor(s, o, 32);
    out[(long long)row * NCLS + lane] = v - m - __logf(s);
}

// ---------------- launch ----------------

extern "C" void kernel_launch(void* const* d_in, const int* in_sizes, int n_in,
                              void* d_out, int out_size, void* d_ws, size_t ws_size,
                              hipStream_t stream) {
    const float* x     = (const float*)d_in[0];
    const int*   ei    = (const int*)d_in[1];
    const float* W1    = (const float*)d_in[2];
    const float* b1    = (const float*)d_in[3];
    const float* W2    = (const float*)d_in[4];
    const float* b2    = (const float*)d_in[5];
    const float* gamma = (const float*)d_in[6];
    float*       out   = (float*)d_out;

    const int nN = in_sizes[0] / F_IN;   // 100000
    const int nE = in_sizes[1] / 2;      // 3200000
    const long long nF = (long long)nN * NCLS;

    // carve workspace
    char* ws = (char*)d_ws;
    size_t off = 0;
    auto carve = [&](size_t bytes) -> void* {
        void* p = ws + off;
        off += (bytes + 255) & ~(size_t)255;
        return p;
    };
    float*  deg   = (float*)carve((size_t)nN * 4);
    float*  dinv  = (float*)carve((size_t)nN * 4);
    float*  norm  = (float*)carve((size_t)nE * 4);
    float*  h     = (float*)carve((size_t)nF * 4);
    float*  hk    = (float*)carve((size_t)nF * 4);
    float*  hn    = (float*)carve((size_t)nF * 4);
    float*  accb  = (float*)carve((size_t)nF * 4);
    __bf16* w1p   = (__bf16*)carve((size_t)F_IN * HID * 2);
    __bf16* w2p   = (__bf16*)carve((size_t)HID * NCLS * 2);
    (void)ws_size; (void)n_in; (void)out_size;

    // GCN normalization (deg init 1.0 accounts for self-loops)
    fill_f32<<<(nN + 255) / 256, 256, 0, stream>>>(deg, 1.0f, nN);
    degree_kernel<<<(nE + 255) / 256, 256, 0, stream>>>(ei, deg, nE);
    dinv_kernel<<<(nN + 255) / 256, 256, 0, stream>>>(deg, dinv, nN);
    norm_kernel<<<(nE + 255) / 256, 256, 0, stream>>>(ei, dinv, norm, nE);

    // weights -> fragment-major bf16
    pack_b_frag<<<(F_IN * HID + 255) / 256, 256, 0, stream>>>(W1, w1p, F_IN, HID);
    pack_b_frag<<<(HID * NCLS + 255) / 256, 256, 0, stream>>>(W2, w2p, HID, NCLS);

    // fused MLP (WMMA bf16, f32 accumulate)
    mlp_fused<<<(nN + 15) / 16, 128, 0, stream>>>(x, w1p, b1, w2p, b2, h, nN);

    // propagation
    init_acc_kernel<<<(int)((nF + 255) / 256), 256, 0, stream>>>(h, hk, accb, gamma, nF);
    float* cur = hk;
    float* nxt = hn;
    long long edgeThreads = (long long)nE * 32;
    for (int k = 1; k <= KHOP; ++k) {
        fill_f32<<<(int)((nF + 255) / 256), 256, 0, stream>>>(nxt, 0.0f, nF);
        edge_prop_kernel<<<(int)((edgeThreads + 255) / 256), 256, 0, stream>>>(ei, norm, cur, nxt, nE);
        finish_prop_kernel<<<(int)((nF + 255) / 256), 256, 0, stream>>>(nxt, cur, dinv, accb, gamma, k, nF);
        float* tmp = cur; cur = nxt; nxt = tmp;
    }

    logsoftmax_kernel<<<(nN + 7) / 8, 256, 0, stream>>>(accb, out, nN);
}
